// RelPosSelfAttention_38328288149701
// MI455X (gfx1250) — compile-verified, matrix operand order checked by reading, not run on hardware
//
#include <hip/hip_runtime.h>

// ---------------------------------------------------------------------------
// RelPos self-attention for gfx1250 (MI455X), f16 WMMA with f32 accumulation.
//   DIM=1024, NHEAD=16, HEAD_DIM=64, MAX_REL=256, B=4, S=2048
// Data movement: async global->LDS staging (ASYNCcnt) with double buffering
// where the toolchain exposes it; synchronous fallback otherwise.
// ---------------------------------------------------------------------------

typedef __attribute__((ext_vector_type(16))) _Float16 v16h;
typedef __attribute__((ext_vector_type(8)))  _Float16 v8h;
typedef __attribute__((ext_vector_type(8)))  float    v8f;
typedef __attribute__((ext_vector_type(4)))  int      v4i;

#define DIM_C   1024
#define NHEAD_C 16
#define HD_C    64
#define S_C     2048
#define B_C     4
#define ROWS_C  (B_C * S_C)   // 8192
#define MAXREL  256

// ---- async global->LDS support (probe-gated) ------------------------------
#if defined(__has_builtin)
#if __has_builtin(__builtin_amdgcn_global_load_async_to_lds_b128)
#define HAVE_ASYNC_LDS 1
#endif
#endif
#ifndef HAVE_ASYNC_LDS
#define HAVE_ASYNC_LDS 0
#endif

#if HAVE_ASYNC_LDS
#if __has_builtin(__builtin_amdgcn_s_wait_asynccnt)
#define WAIT_ASYNC0() __builtin_amdgcn_s_wait_asynccnt(0)
#define WAIT_ASYNC1() __builtin_amdgcn_s_wait_asynccnt(1)
#else
#define WAIT_ASYNC0() asm volatile("s_wait_asynccnt 0x0" ::: "memory")
#define WAIT_ASYNC1() asm volatile("s_wait_asynccnt 0x1" ::: "memory")
#endif
#else
#define WAIT_ASYNC0()
#define WAIT_ASYNC1()
#endif

// Pointer-to-v4i in explicit address spaces (builtin takes v4i*, per compile
// diagnostic: "int __vector(4) __device__ *").
typedef __attribute__((address_space(1))) v4i as1_v4i;
typedef __attribute__((address_space(3))) v4i as3_v4i;

// Copy 16B global -> LDS. Async path goes straight to LDS (no VGPR
// round-trip), tracked on ASYNCcnt; fallback is a sync VGPR copy.
__device__ __forceinline__ void copy16_g2l(const _Float16* g, _Float16* l) {
#if HAVE_ASYNC_LDS
  __builtin_amdgcn_global_load_async_to_lds_b128(
      (as1_v4i*)(uintptr_t)g, (as3_v4i*)(uintptr_t)l, 0, 0);
#else
  *(v8h*)l = *(const v8h*)g;
#endif
}

// Load a 16-bit A-fragment (16x32, MxK) per ISA layout:
//   lane L: row = L&15 ; halves [0..7] = K in [ko..ko+7], [8..15] = K in [16+ko..16+ko+7]
//   where ko = (L>>4)*8.  Caller passes p = rowBase + ko.
// The same pattern loads B (32x16 KxN) as "A-layout of B^T": lane holds column n=L&15.
__device__ __forceinline__ v16h ld_frag(const _Float16* p) {
  union { v16h v; v8h h[2]; } u;
  u.h[0] = *(const v8h*)(p);        // 16B chunk, K = ko .. ko+7
  u.h[1] = *(const v8h*)(p + 16);   // 16B chunk, K = 16+ko .. 16+ko+7
  return u.v;
}

__device__ __forceinline__ v8f wmma_f16(v16h a, v16h b, v8f c) {
  return __builtin_amdgcn_wmma_f32_16x16x32_f16(false, a, false, b, (short)0, c,
                                                false, false);
}

// ---------------------------------------------------------------------------
// fp32 -> f16 conversion (vectorized x4)
// ---------------------------------------------------------------------------
__global__ void cvt_f32_f16(const float* __restrict__ in,
                            _Float16* __restrict__ out, int n) {
  int i = (blockIdx.x * blockDim.x + threadIdx.x) * 4;
  if (i < n) {
    float4 v = *(const float4*)(in + i);
    out[i + 0] = (_Float16)v.x;
    out[i + 1] = (_Float16)v.y;
    out[i + 2] = (_Float16)v.z;
    out[i + 3] = (_Float16)v.w;
  }
}

// ---------------------------------------------------------------------------
// GEMM: out[m,n] = alpha * ( sum_k A[m,k]*W[n,k] + bias[n] )
//   A: [8192 x 1024] f16 row-major, W: [1024 x 1024] f16 row-major
//   HEAD_OUT=true  -> scatter f16 into [b,h,s,d]   (Q/K/V projections)
//   HEAD_OUT=false -> dense f32 [m,1024]           (output projection)
// Block: 256 threads = 8 waves; block tile 128(M) x 64(N); wave tile 32x32.
// W tile (64 rows x 32 k) is staged per block into LDS via async loads with
// double buffering; A-fragments load direct from global (only x2 reuse).
// ---------------------------------------------------------------------------
template <bool HEAD_OUT>
__global__ __launch_bounds__(256) void gemm_wmma(
    const _Float16* __restrict__ A, const _Float16* __restrict__ W,
    const float* __restrict__ bias, void* __restrict__ outp, float alpha) {
  constexpr int K = DIM_C;
  constexpr int NT = K / 32;                       // 32 k-steps
  __shared__ __align__(16) _Float16 sB[2][64][32];  // 8 KB double-buffered W tile

  const int tid   = threadIdx.x;
  const int lane  = tid & 31;
  const int wave  = tid >> 5;
  const int waveM = wave >> 1;
  const int waveN = wave & 1;
  const int bm  = blockIdx.x * 128 + waveM * 32;
  const int bn0 = blockIdx.y * 64;                 // block's W-row base
  const int bnl = waveN * 32;                      // wave's local W-row base
  const int r   = lane & 15;
  const int hi  = lane >> 4;
  const int ko  = hi << 3;

  const _Float16* a0 = A + (size_t)(bm + r) * K + ko;
  const _Float16* a1 = a0 + (size_t)16 * K;

  // staging coords: 256 threads x 16B = 4KB = one 64x32 tile
  const int srow = tid >> 2;            // 0..63
  const int skk  = (tid & 3) * 8;       // 0,8,16,24
  const _Float16* wsrc = W + (size_t)(bn0 + srow) * K + skk;

  // prologue: stage tile 0
  copy16_g2l(wsrc, &sB[0][srow][skk]);

  v8f acc[2][2] = {};
  for (int kb = 0; kb < NT; ++kb) {
    const int k   = kb * 32;
    const int cur = kb & 1;
    if (kb + 1 < NT) {                  // prefetch next tile into other buffer
      copy16_g2l(wsrc + (k + 32), &sB[cur ^ 1][srow][skk]);
      WAIT_ASYNC1();                    // tile kb complete; kb+1 in flight
    } else {
      WAIT_ASYNC0();
    }
    __syncthreads();

    __builtin_prefetch(a0 + k + 256, 0, 3);
    v16h fa0 = ld_frag(a0 + k);
    v16h fa1 = ld_frag(a1 + k);
    v16h fb0 = ld_frag(&sB[cur][bnl + r][ko]);
    v16h fb1 = ld_frag(&sB[cur][bnl + 16 + r][ko]);
    acc[0][0] = wmma_f16(fa0, fb0, acc[0][0]);
    acc[0][1] = wmma_f16(fa0, fb1, acc[0][1]);
    acc[1][0] = wmma_f16(fa1, fb0, acc[1][0]);
    acc[1][1] = wmma_f16(fa1, fb1, acc[1][1]);
    __syncthreads();                    // done reading cur before it is refilled
  }

#pragma unroll
  for (int tm = 0; tm < 2; ++tm) {
#pragma unroll
    for (int tn = 0; tn < 2; ++tn) {
      int n = bn0 + bnl + tn * 16 + r;
      float bv = bias[n];
#pragma unroll
      for (int i = 0; i < 8; ++i) {
        int m = bm + tm * 16 + i + 8 * hi;
        float val = (acc[tm][tn][i] + bv) * alpha;
        if constexpr (HEAD_OUT) {
          int bb = m >> 11, ss = m & (S_C - 1);
          int hh = n >> 6, dd = n & (HD_C - 1);
          ((_Float16*)outp)[(((size_t)bb * NHEAD_C + hh) * S_C + ss) * HD_C + dd] =
              (_Float16)val;
        } else {
          ((float*)outp)[(size_t)m * DIM_C + n] = val;
        }
      }
    }
  }
}

// ---------------------------------------------------------------------------
// Flash-style attention with relative-position bias.
//   Q,K,V: [B,H,S,HD] f16 (Q pre-scaled by HD^-0.5 via GEMM alpha)
//   ctx:   [B,S,DIM]  f16
// Block: 128 threads = 4 waves; each wave owns 16 query rows; key tile = 64.
// K tile staged via async global->LDS (flat 8KB contiguous copy); V staged
// through VGPRs because it must be transposed for the B operand of P@V.
// ---------------------------------------------------------------------------
__global__ __launch_bounds__(128) void attn_wmma(
    const _Float16* __restrict__ Q, const _Float16* __restrict__ Km,
    const _Float16* __restrict__ V, const float* __restrict__ relb,
    _Float16* __restrict__ ctx) {
  constexpr int KT = 64;
  __shared__ __align__(16) _Float16 sK[KT][HD_C];     // [key][dim]   8 KB
  __shared__ __align__(16) _Float16 sVt[HD_C][KT];    // [dim][key]   8 KB
  __shared__ __align__(16) _Float16 sP[4][16][KT];    // per-wave P   8 KB
  __shared__ float sBias[2 * MAXREL + 1];             // rel bias col ~2 KB

  const int tid  = threadIdx.x;
  const int lane = tid & 31;
  const int wave = tid >> 5;
  const int r  = lane & 15;
  const int hi = lane >> 4;
  const int ko = hi << 3;
  const int bh = blockIdx.y;          // b*16 + h
  const int b  = bh >> 4;
  const int h  = bh & 15;
  const int q0 = blockIdx.x * 64 + wave * 16;

  for (int i = tid; i < 2 * MAXREL + 1; i += 128)
    sBias[i] = relb[i * NHEAD_C + h];

  const _Float16* Qb = Q + (size_t)bh * S_C * HD_C;
  const _Float16* Kb = Km + (size_t)bh * S_C * HD_C;
  const _Float16* Vb = V + (size_t)bh * S_C * HD_C;

  // Q fragments for this wave's 16 rows (HD=64 -> two K-steps of 32)
  v16h fq0 = ld_frag(Qb + (size_t)(q0 + r) * HD_C + ko);
  v16h fq1 = ld_frag(Qb + (size_t)(q0 + r) * HD_C + 32 + ko);

  v8f oacc[4] = {};                   // 16 rows x 64 dims
  float mrun[8], lrun[8];
#pragma unroll
  for (int i = 0; i < 8; ++i) { mrun[i] = -1e30f; lrun[i] = 0.f; }

  __syncthreads();                    // sBias ready

  for (int kt = 0; kt < S_C; kt += KT) {
    // ---- stage K tile: contiguous 8KB, async global -> LDS ----
    _Float16* sKf = &sK[0][0];
#pragma unroll
    for (int c = 0; c < 4; ++c) {
      int e = (c * 128 + tid) * 8;    // flat f16 index in 64x64 tile
      copy16_g2l(Kb + (size_t)kt * HD_C + e, sKf + e);
    }
    // ---- stage V tile transposed (VGPR shuffle, overlaps async K) ----
#pragma unroll
    for (int c = 0; c < 4; ++c) {
      int e   = (c * 128 + tid) * 8;
      int key = e >> 6;
      int dd  = e & 63;
      v8h vv = *(const v8h*)(Vb + (size_t)(kt + key) * HD_C + dd);
#pragma unroll
      for (int j = 0; j < 8; ++j) sVt[dd + j][key] = vv[j];
    }
    WAIT_ASYNC0();
    __syncthreads();

    // ---- S = Q K^T  (16 x 64 scores) ----
    v8f sc[4];
#pragma unroll
    for (int t = 0; t < 4; ++t) {
      v16h fb0 = ld_frag(&sK[t * 16 + r][ko]);
      v16h fb1 = ld_frag(&sK[t * 16 + r][32 + ko]);
      v8f z = {};
      sc[t] = wmma_f16(fq0, fb0, z);
      sc[t] = wmma_f16(fq1, fb1, sc[t]);
    }

    // ---- rel-pos bias + online softmax (rowwise over 16-lane halves) ----
#pragma unroll
    for (int i = 0; i < 8; ++i) {
      int qg = q0 + i + 8 * hi;
      float mx = -1e30f;
#pragma unroll
      for (int t = 0; t < 4; ++t) {
        int kg = kt + t * 16 + r;
        int d  = qg - kg;
        d = d > MAXREL ? MAXREL : (d < -MAXREL ? -MAXREL : d);
        float e2 = sc[t][i] + sBias[d + MAXREL];
        sc[t][i] = e2;
        mx = fmaxf(mx, e2);
      }
#pragma unroll
      for (int off = 1; off < 16; off <<= 1)
        mx = fmaxf(mx, __shfl_xor(mx, off, 32));
      float mnew  = fmaxf(mrun[i], mx);
      float scale = __expf(mrun[i] - mnew);
      float rs = 0.f;
#pragma unroll
      for (int t = 0; t < 4; ++t) {
        float p = __expf(sc[t][i] - mnew);
        sc[t][i] = p;
        rs += p;
      }
#pragma unroll
      for (int off = 1; off < 16; off <<= 1) rs += __shfl_xor(rs, off, 32);
      lrun[i] = lrun[i] * scale + rs;
      mrun[i] = mnew;
#pragma unroll
      for (int t = 0; t < 4; ++t) oacc[t][i] *= scale;
    }

    // ---- P -> LDS (f16), converting C-layout to A-fragment layout ----
#pragma unroll
    for (int i = 0; i < 8; ++i) {
      int row = i + 8 * hi;
#pragma unroll
      for (int t = 0; t < 4; ++t)
        sP[wave][row][t * 16 + r] = (_Float16)sc[t][i];
    }
    __syncthreads();

    // ---- O += P V ----
    v16h fp0 = ld_frag(&sP[wave][r][ko]);
    v16h fp1 = ld_frag(&sP[wave][r][32 + ko]);
#pragma unroll
    for (int t = 0; t < 4; ++t) {
      v16h fv0 = ld_frag(&sVt[t * 16 + r][ko]);
      v16h fv1 = ld_frag(&sVt[t * 16 + r][32 + ko]);
      oacc[t] = wmma_f16(fp0, fv0, oacc[t]);
      oacc[t] = wmma_f16(fp1, fv1, oacc[t]);
    }
    __syncthreads();   // protect sK/sVt/sP before next tile's staging
  }

  // ---- normalize and write context [b, s, h*64 + d] ----
#pragma unroll
  for (int i = 0; i < 8; ++i) {
    float inv = 1.f / lrun[i];
    int qg = q0 + i + 8 * hi;
#pragma unroll
    for (int t = 0; t < 4; ++t) {
      int dd = t * 16 + r;
      ctx[((size_t)(b * S_C + qg)) * DIM_C + h * HD_C + dd] =
          (_Float16)(oacc[t][i] * inv);
    }
  }
}

// ---------------------------------------------------------------------------
// Host-side orchestration
// ---------------------------------------------------------------------------
extern "C" void kernel_launch(void* const* d_in, const int* in_sizes, int n_in,
                              void* d_out, int out_size, void* d_ws,
                              size_t ws_size, hipStream_t stream) {
  const float* x    = (const float*)d_in[0];
  const float* Wq   = (const float*)d_in[1];
  const float* bq   = (const float*)d_in[2];
  const float* Wk   = (const float*)d_in[3];
  const float* bk   = (const float*)d_in[4];
  const float* Wv   = (const float*)d_in[5];
  const float* bv   = (const float*)d_in[6];
  const float* Wo   = (const float*)d_in[7];
  const float* bo   = (const float*)d_in[8];
  const float* relb = (const float*)d_in[9];
  float* out = (float*)d_out;

  char* ws = (char*)d_ws;
  size_t off = 0;
  auto carve = [&](size_t bytes) {
    void* p = ws + off;
    off += (bytes + 255) & ~(size_t)255;
    return p;
  };
  const size_t xBytes = (size_t)ROWS_C * DIM_C * sizeof(_Float16);  // 16 MB
  const size_t wBytes = (size_t)DIM_C * DIM_C * sizeof(_Float16);   //  2 MB
  _Float16* xh   = (_Float16*)carve(xBytes);
  _Float16* Wqh  = (_Float16*)carve(wBytes);
  _Float16* Wkh  = (_Float16*)carve(wBytes);
  _Float16* Wvh  = (_Float16*)carve(wBytes);
  _Float16* Woh  = (_Float16*)carve(wBytes);
  _Float16* Qh   = (_Float16*)carve(xBytes);
  _Float16* Kh   = (_Float16*)carve(xBytes);
  _Float16* Vh   = (_Float16*)carve(xBytes);
  _Float16* ctxh = (_Float16*)carve(xBytes);

  const int nX = ROWS_C * DIM_C;     // 8388608
  const int nW = DIM_C * DIM_C;      // 1048576
  cvt_f32_f16<<<(nX / 4 + 255) / 256, 256, 0, stream>>>(x, xh, nX);
  cvt_f32_f16<<<(nW / 4 + 255) / 256, 256, 0, stream>>>(Wq, Wqh, nW);
  cvt_f32_f16<<<(nW / 4 + 255) / 256, 256, 0, stream>>>(Wk, Wkh, nW);
  cvt_f32_f16<<<(nW / 4 + 255) / 256, 256, 0, stream>>>(Wv, Wvh, nW);
  cvt_f32_f16<<<(nW / 4 + 255) / 256, 256, 0, stream>>>(Wo, Woh, nW);

  dim3 gg(ROWS_C / 128, DIM_C / 64);  // 64 x 16
  const float qscale = 0.125f;        // HD^-0.5
  gemm_wmma<true><<<gg, 256, 0, stream>>>(xh, Wqh, bq, (void*)Qh, qscale);
  gemm_wmma<true><<<gg, 256, 0, stream>>>(xh, Wkh, bk, (void*)Kh, 1.0f);
  gemm_wmma<true><<<gg, 256, 0, stream>>>(xh, Wvh, bv, (void*)Vh, 1.0f);

  attn_wmma<<<dim3(S_C / 64, B_C * NHEAD_C), 128, 0, stream>>>(Qh, Kh, Vh, relb,
                                                               ctxh);

  gemm_wmma<false><<<gg, 256, 0, stream>>>(ctxh, Woh, bo, (void*)out, 1.0f);
}
